// FFM_31825707663795
// MI455X (gfx1250) — matrix operand order repaired and checked by reading, not compiled, not run on hardware
//
#include <hip/hip_runtime.h>

// ---------------------------------------------------------------------------
// FFM block for MI455X (gfx1250, wave32, WMMA bf16 16x16x32, f32 accumulate)
// Chunked parallel scan + permuted-K z_in layout + LDS-free big GEMM.
// ---------------------------------------------------------------------------

#define T_    4096
#define D_    128
#define HID_  64
#define TR_   64
#define CT_   64
#define K2_   128     // 2*TRACE
#define KZ_   8192    // 2*TRACE*CONTEXT
#define OUT_  128
#define NCH_  64      // scan chunks
#define CHL_  64      // chunk length (NCH_*CHL_ == T_)

typedef __bf16  v16bf __attribute__((ext_vector_type(16)));
typedef __bf16  v8bf  __attribute__((ext_vector_type(8)));
typedef __bf16  v2bf  __attribute__((ext_vector_type(2)));
typedef float   v8f   __attribute__((ext_vector_type(8)));

__device__ __forceinline__ float sigmoidf_(float x) {
    return 1.0f / (1.0f + __expf(-x));
}

// A-fragment (16x32 bf16): lane L<16 = row L, K = 0..15;
// lane L>=16 = row L-16, K = 16..31. Two contiguous 16B chunks.
__device__ __forceinline__ v16bf mk_afrag(const __bf16* rowk /* &tile[row][kt*32] */) {
    const int base = (threadIdx.x & 16) ? 16 : 0;
    v8bf lo = *(const v8bf*)(rowk + base);
    v8bf hi = *(const v8bf*)(rowk + base + 8);
    v16bf a;
#pragma unroll
    for (int i = 0; i < 8; ++i) { a[i] = lo[i]; a[8 + i] = hi[i]; }
    return a;
}

// B-fragment (32x16 bf16): lane L<16 = col L, K=0..15; lane L>=16 = col L-16,
// K=16..31. With B staged transposed [n][k] this is one 32B contiguous read.
__device__ __forceinline__ v16bf mk_bfrag(const __bf16* colk /* &Wt[n][kt*32] */) {
    const int khalf = (threadIdx.x & 16) ? 16 : 0;
    return *(const v16bf*)(colk + khalf);
}

__device__ __forceinline__ v8f wmma_bf16(v16bf a, v16bf b, v8f c) {
    return __builtin_amdgcn_wmma_f32_16x16x32_bf16(
        /*neg_a=*/false, a, /*neg_b=*/false, b,
        /*c_mod=*/(short)0, c, /*reuse_a=*/false, /*reuse_b=*/false);
}

// ---------------------------------------------------------------------------
// Kernel 0: one-time Wz prep. Store Wz as bf16 in B-operand layout [n][k],
// with K permuted to match the scan's packed (re,im) emission order:
//   k = 2*(r*64+c) + p  <->  original row = r*128 + c + p*64.
// ---------------------------------------------------------------------------
__global__ __launch_bounds__(256) void k0_wzprep(
    const float* __restrict__ Wz, __bf16* __restrict__ wzb)
{
    int e = blockIdx.x * 256 + threadIdx.x;      // 0 .. 64*8192-1
    int n = e >> 13;
    int k = e & (KZ_ - 1);
    int p = k & 1, pair = k >> 1;
    int r = pair >> 6, c = pair & 63;
    int orow = r * 128 + c + p * 64;
    wzb[(size_t)n * KZ_ + k] = (__bf16)Wz[(size_t)orow * 64 + n];
}

// ---------------------------------------------------------------------------
// Kernel 1: front-end GEMM chain. 64 rows of T per block, 128 threads (4 waves),
// each wave owns a 16-row strip. All GEMMs via v_wmma_f32_16x16x32_bf16.
// ---------------------------------------------------------------------------
__global__ __launch_bounds__(128) void k1_frontend(
    const float* __restrict__ x,
    const float* __restrict__ W1,  const float* __restrict__ b1,
    const float* __restrict__ W2,  const float* __restrict__ b2,
    const float* __restrict__ Wgi, const float* __restrict__ bgi,
    const float* __restrict__ Wp,  const float* __restrict__ bp,
    const float* __restrict__ Wgo, const float* __restrict__ bgo,
    const float* __restrict__ Ws,  const float* __restrict__ bs,
    float* __restrict__ g_gated, float* __restrict__ g_go, float* __restrict__ g_skip)
{
    __shared__ __align__(32) __bf16 buf1[64][128];   // x tile, later h2 (bf16)
    __shared__ __align__(32) __bf16 buf2[64][128];   // W1^T, later gate weights^T
    __shared__ __align__(32) __bf16 w2t[128][64];    // W2^T  [n][k]
    __shared__ __align__(32) __bf16 hbuf[64][64];    // h (bf16)

    const int tid  = threadIdx.x;
    const int w    = tid >> 5;
    const int lane = tid & 31;
    const int r0   = blockIdx.x * 64;

    const int arow = (lane & 15) + w * 16;
    const int ncol = lane & 15;
    const int mof  = (tid & 16) ? 8 : 0;

    // ---- cooperative loads: x tile, W1^T, W2^T ----------------------------
#pragma unroll
    for (int i = 0; i < 64; ++i) {
        int e = tid + i * 128;
        int rr = e >> 7, cc = e & 127;
        buf1[rr][cc] = (__bf16)x[(size_t)(r0 + rr) * D_ + cc];
    }
#pragma unroll
    for (int i = 0; i < 64; ++i) {
        int e = tid + i * 128;
        int n = e & 63, k = e >> 6;
        buf2[n][k] = (__bf16)W1[(size_t)k * 64 + n];
    }
#pragma unroll
    for (int i = 0; i < 64; ++i) {
        int e = tid + i * 128;
        int n = e & 127, k = e >> 7;
        w2t[n][k] = (__bf16)W2[(size_t)k * 128 + n];
    }
    __syncthreads();

    // ---- h = relu(x @ W1 + b1) --------------------------------------------
    {
        v8f acc[4] = {};
#pragma unroll
        for (int kt = 0; kt < 4; ++kt) {
            v16bf a = mk_afrag(&buf1[arow][kt * 32]);
#pragma unroll
            for (int nt = 0; nt < 4; ++nt) {
                v16bf b = mk_bfrag(&buf2[ncol + nt * 16][kt * 32]);
                acc[nt] = wmma_bf16(a, b, acc[nt]);
            }
        }
#pragma unroll
        for (int nt = 0; nt < 4; ++nt) {
            int n = ncol + nt * 16;
            float bias = b1[n];
#pragma unroll
            for (int v = 0; v < 8; ++v) {
                float val = acc[nt][v] + bias;
                hbuf[w * 16 + mof + v][n] = (__bf16)(val > 0.f ? val : 0.f);
            }
        }
    }
    __syncthreads();

    // ---- h2 = h @ W2 + b2  -> buf1 (bf16) ---------------------------------
    {
        v8f acc[8] = {};
#pragma unroll
        for (int kt = 0; kt < 2; ++kt) {
            v16bf a = mk_afrag(&hbuf[arow][kt * 32]);
#pragma unroll
            for (int nt = 0; nt < 8; ++nt) {
                v16bf b = mk_bfrag(&w2t[ncol + nt * 16][kt * 32]);
                acc[nt] = wmma_bf16(a, b, acc[nt]);
            }
        }
#pragma unroll
        for (int nt = 0; nt < 8; ++nt) {
            int n = ncol + nt * 16;
            float bias = b2[n];
#pragma unroll
            for (int v = 0; v < 8; ++v)
                buf1[w * 16 + mof + v][n] = (__bf16)(acc[nt][v] + bias);
        }
    }
    __syncthreads();

    // ---- gate GEMMs off h2: K=128, N=64 each ------------------------------
    auto load_gate_w = [&](const float* __restrict__ W) {
#pragma unroll
        for (int i = 0; i < 64; ++i) {
            int e = tid + i * 128;
            int n = e & 63, k = e >> 6;
            buf2[n][k] = (__bf16)W[(size_t)k * 64 + n];
        }
    };
    auto gate_gemm = [&](v8f acc[4]) {
#pragma unroll
        for (int kt = 0; kt < 4; ++kt) {
            v16bf a = mk_afrag(&buf1[arow][kt * 32]);
#pragma unroll
            for (int nt = 0; nt < 4; ++nt) {
                v16bf b = mk_bfrag(&buf2[ncol + nt * 16][kt * 32]);
                acc[nt] = wmma_bf16(a, b, acc[nt]);
            }
        }
    };

    // gate_in = sigmoid(h2 @ Wgi + bgi)  (kept in registers)
    load_gate_w(Wgi);
    __syncthreads();
    v8f gi[4] = {};
    gate_gemm(gi);
#pragma unroll
    for (int nt = 0; nt < 4; ++nt) {
        float bias = bgi[ncol + nt * 16];
#pragma unroll
        for (int v = 0; v < 8; ++v) gi[nt][v] = sigmoidf_(gi[nt][v] + bias);
    }
    __syncthreads();

    // gated = sigmoid(h2 @ Wp + bp) * gate_in  -> global
    load_gate_w(Wp);
    __syncthreads();
    {
        v8f pr[4] = {};
        gate_gemm(pr);
#pragma unroll
        for (int nt = 0; nt < 4; ++nt) {
            int n = ncol + nt * 16;
            float bias = bp[n];
#pragma unroll
            for (int v = 0; v < 8; ++v) {
                float g = sigmoidf_(pr[nt][v] + bias) * gi[nt][v];
                g_gated[(size_t)(r0 + w * 16 + mof + v) * TR_ + n] = g;
            }
        }
    }
    __syncthreads();

    // gate_out = sigmoid(h2 @ Wgo + bgo) -> global
    load_gate_w(Wgo);
    __syncthreads();
    {
        v8f go[4] = {};
        gate_gemm(go);
#pragma unroll
        for (int nt = 0; nt < 4; ++nt) {
            int n = ncol + nt * 16;
            float bias = bgo[n];
#pragma unroll
            for (int v = 0; v < 8; ++v)
                g_go[(size_t)(r0 + w * 16 + mof + v) * HID_ + n] =
                    sigmoidf_(go[nt][v] + bias);
        }
    }
    __syncthreads();

    // skip = h2 @ Ws + bs -> global
    load_gate_w(Ws);
    __syncthreads();
    {
        v8f sk[4] = {};
        gate_gemm(sk);
#pragma unroll
        for (int nt = 0; nt < 4; ++nt) {
            int n = ncol + nt * 16;
            float bias = bs[n];
#pragma unroll
            for (int v = 0; v < 8; ++v)
                g_skip[(size_t)(r0 + w * 16 + mof + v) * HID_ + n] =
                    sk[nt][v] + bias;
        }
    }
}

// ---------------------------------------------------------------------------
// Kernel 2a: chunk aggregates. Grid = 64 chunks x 16 row-groups, 256 threads
// (one per (r,c) pair, 4 r per block). Local scan from s=0 over 64 steps ->
// B (chunk offset term) and per-chunk "any reset" flag.
// ---------------------------------------------------------------------------
__global__ __launch_bounds__(256) void k2a_aggregate(
    const float* __restrict__ gated, const unsigned char* __restrict__ start,
    const float* __restrict__ a, const float* __restrict__ b,
    float* __restrict__ agg, int* __restrict__ rst)
{
    __shared__ float         gLDS[CHL_][4];
    __shared__ unsigned char sLDS[CHL_];

    const int tid   = threadIdx.x;
    const int chunk = blockIdx.x >> 4;
    const int rb    = blockIdx.x & 15;
    const int lr    = tid >> 6;
    const int c     = tid & 63;
    const int r     = rb * 4 + lr;
    const int idx   = r * CT_ + c;
    const int t0    = chunk * CHL_;

    { int tt = tid >> 2, rr = tid & 3;
      gLDS[tt][rr] = gated[(size_t)(t0 + tt) * TR_ + rb * 4 + rr]; }
    if (tid < CHL_) sLDS[tid] = start[t0 + tid];
    __syncthreads();

    const float dr = __expf(-fabsf(a[r]));
    const float cr = dr * __cosf(b[c]);
    const float ci = dr * __sinf(b[c]);

    float sre = 0.f, sim = 0.f;
    int any = 0;
    for (int t = 0; t < CHL_; ++t) {
        float g  = gLDS[t][lr];
        int   st = sLDS[t];
        any |= st;
        float keep = st ? 0.f : 1.f;
        float nre = keep * (sre * cr - sim * ci) + g;
        float nim = keep * (sre * ci + sim * cr);
        sre = nre; sim = nim;
    }
    size_t o = ((size_t)chunk * 4096 + idx) * 2;
    agg[o] = sre; agg[o + 1] = sim;
    if (tid == 0) rst[chunk] = any;   // same value from all 16 row-groups
}

// ---------------------------------------------------------------------------
// Kernel 2b: sequential prefix over the 64 chunks (4096 parallel recurrences).
// s_{i+1} = rst_i ? B_i : gamma^64 * s_i + B_i.  Also writes the final state.
// ---------------------------------------------------------------------------
__global__ __launch_bounds__(256) void k2b_prefix(
    const float* __restrict__ state0, const float* __restrict__ agg,
    const int* __restrict__ rst,
    const float* __restrict__ a, const float* __restrict__ b,
    float* __restrict__ init, float* __restrict__ out_state)
{
    const int idx = blockIdx.x * 256 + threadIdx.x;   // 0..4095
    const int r = idx >> 6, c = idx & 63;

    float sre = state0[2 * idx];
    float sim = state0[2 * idx + 1];

    const float dr  = expf(-(float)CHL_ * fabsf(a[r]));
    const float ang = (float)CHL_ * b[c];
    const float cr  = dr * cosf(ang);
    const float ci  = dr * sinf(ang);

    for (int i = 0; i < NCH_; ++i) {
        size_t o = ((size_t)i * 4096 + idx) * 2;
        init[o] = sre; init[o + 1] = sim;
        float bre = agg[o], bim = agg[o + 1];
        if (rst[i]) { sre = bre; sim = bim; }
        else {
            float nre = sre * cr - sim * ci + bre;
            float nim = sre * ci + sim * cr + bim;
            sre = nre; sim = nim;
        }
    }
    out_state[2 * idx]     = sre;
    out_state[2 * idx + 1] = sim;
}

// ---------------------------------------------------------------------------
// Kernel 2c: emit pass. Re-run each chunk from its correct initial state and
// write z_in rows as packed (re,im) bf16 dwords: column k = 2*(r*64+c)+{0,1}.
// ---------------------------------------------------------------------------
__global__ __launch_bounds__(256) void k2c_emit(
    const float* __restrict__ gated, const unsigned char* __restrict__ start,
    const float* __restrict__ a, const float* __restrict__ b,
    const float* __restrict__ init, __bf16* __restrict__ zin)
{
    __shared__ float         gLDS[CHL_][4];
    __shared__ unsigned char sLDS[CHL_];

    const int tid   = threadIdx.x;
    const int chunk = blockIdx.x >> 4;
    const int rb    = blockIdx.x & 15;
    const int lr    = tid >> 6;
    const int c     = tid & 63;
    const int r     = rb * 4 + lr;
    const int idx   = r * CT_ + c;
    const int t0    = chunk * CHL_;

    { int tt = tid >> 2, rr = tid & 3;
      gLDS[tt][rr] = gated[(size_t)(t0 + tt) * TR_ + rb * 4 + rr]; }
    if (tid < CHL_) sLDS[tid] = start[t0 + tid];
    __syncthreads();

    const float dr = __expf(-fabsf(a[r]));
    const float cr = dr * __cosf(b[c]);
    const float ci = dr * __sinf(b[c]);

    size_t o = ((size_t)chunk * 4096 + idx) * 2;
    float sre = init[o], sim = init[o + 1];

    for (int t = 0; t < CHL_; ++t) {
        float g    = gLDS[t][lr];
        float keep = sLDS[t] ? 0.f : 1.f;
        float nre  = keep * (sre * cr - sim * ci) + g;
        float nim  = keep * (sre * ci + sim * cr);
        sre = nre; sim = nim;
        v2bf pk; pk[0] = (__bf16)sre; pk[1] = (__bf16)sim;
        *(v2bf*)(zin + (size_t)(t0 + t) * KZ_ + 2 * idx) = pk;   // dword store
    }
}

// ---------------------------------------------------------------------------
// Kernel 3: z = z_in(4096x8192 bf16) @ Wz_bf16 + bz. LDS-free: A fragments are
// two b128 global loads/lane, B fragments are 32B loads from the prepped
// [n][k] bf16 copy (L2/WGP$-resident). Prefetch the A tile 8 k-steps ahead.
// ---------------------------------------------------------------------------
__global__ __launch_bounds__(128) void k3_zgemm(
    const __bf16* __restrict__ zin, const __bf16* __restrict__ wzb,
    const float* __restrict__ bz, float* __restrict__ zout)
{
    const int tid   = threadIdx.x;
    const int w     = tid >> 5;
    const int lane  = tid & 31;
    const int ncol  = lane & 15;
    const int mof   = (tid & 16) ? 8 : 0;
    const int khalf = (tid & 16) ? 16 : 0;
    const int row   = blockIdx.x * 64 + w * 16 + (lane & 15);

    const __bf16* arowp = zin + (size_t)row * KZ_;

    v8f acc[4] = {};
    for (int kt = 0; kt < KZ_ / 32; ++kt) {
        v16bf afrag = mk_afrag(arowp + kt * 32);
        if (kt + 8 < KZ_ / 32)
            __builtin_prefetch(arowp + (kt + 8) * 32, 0, 0);
#pragma unroll
        for (int nt = 0; nt < 4; ++nt) {
            v16bf bfrag = *(const v16bf*)(
                wzb + (size_t)(ncol + nt * 16) * KZ_ + kt * 32 + khalf);
            acc[nt] = wmma_bf16(afrag, bfrag, acc[nt]);
        }
    }
#pragma unroll
    for (int nt = 0; nt < 4; ++nt) {
        int n = ncol + nt * 16;
        float bias = bz[n];
#pragma unroll
        for (int v = 0; v < 8; ++v)
            zout[(size_t)(blockIdx.x * 64 + w * 16 + mof + v) * HID_ + n] =
                acc[nt][v] + bias;
    }
}

// ---------------------------------------------------------------------------
// Kernel 4: epilogue. LayerNorm(z*go) + skip*(1-go), then relu(@W3+b3), @W4+b4.
// ---------------------------------------------------------------------------
__global__ __launch_bounds__(128) void k4_epilogue(
    const float* __restrict__ z, const float* __restrict__ go,
    const float* __restrict__ skip,
    const float* __restrict__ W3, const float* __restrict__ b3,
    const float* __restrict__ W4, const float* __restrict__ b4,
    float* __restrict__ out)
{
    __shared__ __align__(32) __bf16 o1[64][64];
    __shared__ __align__(32) __bf16 o2[64][64];
    __shared__ __align__(32) __bf16 w3t[64][64];    // W3^T [n][k]
    __shared__ __align__(32) __bf16 w4t[128][64];   // W4^T [n][k]

    const int tid  = threadIdx.x;
    const int w    = tid >> 5;
    const int lane = tid & 31;
    const int ncol = lane & 15;
    const int mof  = (tid & 16) ? 8 : 0;
    const int arow = (lane & 15) + w * 16;
    const int m0   = blockIdx.x * 64;

    // fused gating + LayerNorm (one thread per row, two passes; L2-hot)
    if (tid < 64) {
        size_t rb = (size_t)(m0 + tid) * HID_;
        float mean = 0.f, m2 = 0.f;
        for (int n = 0; n < HID_; ++n) {
            float u = z[rb + n] * go[rb + n];
            mean += u; m2 += u * u;
        }
        mean *= (1.f / 64.f);
        float var = m2 * (1.f / 64.f) - mean * mean;
        float rs  = rsqrtf(var + 1e-6f);
        for (int n = 0; n < HID_; ++n) {
            float gg = go[rb + n];
            float u  = z[rb + n] * gg;
            float o  = (u - mean) * rs + skip[rb + n] * (1.f - gg);
            o1[tid][n] = (__bf16)o;
        }
    }
#pragma unroll
    for (int i = 0; i < 32; ++i) {                  // W3^T
        int e = tid + i * 128;
        int n = e & 63, k = e >> 6;
        w3t[n][k] = (__bf16)W3[(size_t)k * 64 + n];
    }
#pragma unroll
    for (int i = 0; i < 64; ++i) {                  // W4^T
        int e = tid + i * 128;
        int n = e & 127, k = e >> 7;
        w4t[n][k] = (__bf16)W4[(size_t)k * 128 + n];
    }
    __syncthreads();

    // o2 = relu(o1 @ W3 + b3)
    {
        v8f acc[4] = {};
#pragma unroll
        for (int kt = 0; kt < 2; ++kt) {
            v16bf a = mk_afrag(&o1[arow][kt * 32]);
#pragma unroll
            for (int nt = 0; nt < 4; ++nt) {
                v16bf b = mk_bfrag(&w3t[ncol + nt * 16][kt * 32]);
                acc[nt] = wmma_bf16(a, b, acc[nt]);
            }
        }
#pragma unroll
        for (int nt = 0; nt < 4; ++nt) {
            int n = ncol + nt * 16;
            float bias = b3[n];
#pragma unroll
            for (int v = 0; v < 8; ++v) {
                float val = acc[nt][v] + bias;
                o2[w * 16 + mof + v][n] = (__bf16)(val > 0.f ? val : 0.f);
            }
        }
    }
    __syncthreads();

    // out = o2 @ W4 + b4
    {
        v8f acc[8] = {};
#pragma unroll
        for (int kt = 0; kt < 2; ++kt) {
            v16bf a = mk_afrag(&o2[arow][kt * 32]);
#pragma unroll
            for (int nt = 0; nt < 8; ++nt) {
                v16bf b = mk_bfrag(&w4t[ncol + nt * 16][kt * 32]);
                acc[nt] = wmma_bf16(a, b, acc[nt]);
            }
        }
#pragma unroll
        for (int nt = 0; nt < 8; ++nt) {
            int n = ncol + nt * 16;
            float bias = b4[n];
#pragma unroll
            for (int v = 0; v < 8; ++v)
                out[(size_t)(m0 + w * 16 + mof + v) * OUT_ + n] =
                    acc[nt][v] + bias;
        }
    }
}

// ---------------------------------------------------------------------------
extern "C" void kernel_launch(void* const* d_in, const int* in_sizes, int n_in,
                              void* d_out, int out_size, void* d_ws, size_t ws_size,
                              hipStream_t stream) {
    const float* x      = (const float*)d_in[0];
    const float* state0 = (const float*)d_in[1];   // complex64 interleaved
    const unsigned char* start = (const unsigned char*)d_in[2];
    const float* a   = (const float*)d_in[3];
    const float* b   = (const float*)d_in[4];
    const float* W1  = (const float*)d_in[5];  const float* b1  = (const float*)d_in[6];
    const float* W2  = (const float*)d_in[7];  const float* b2  = (const float*)d_in[8];
    const float* Wgi = (const float*)d_in[9];  const float* bgi = (const float*)d_in[10];
    const float* Wp  = (const float*)d_in[11]; const float* bp  = (const float*)d_in[12];
    const float* Wz  = (const float*)d_in[13]; const float* bz  = (const float*)d_in[14];
    const float* Wgo = (const float*)d_in[15]; const float* bgo = (const float*)d_in[16];
    const float* Ws  = (const float*)d_in[17]; const float* bs  = (const float*)d_in[18];
    const float* W3  = (const float*)d_in[19]; const float* b3  = (const float*)d_in[20];
    const float* W4  = (const float*)d_in[21]; const float* b4  = (const float*)d_in[22];

    // workspace layout (floats unless noted)
    float* ws_gated = (float*)d_ws;                        // T*64
    float* ws_go    = ws_gated + (size_t)T_ * TR_;         // T*64
    float* ws_skip  = ws_go    + (size_t)T_ * HID_;        // T*64
    float* ws_z     = ws_skip  + (size_t)T_ * HID_;        // T*64
    float* ws_agg   = ws_z     + (size_t)T_ * HID_;        // NCH*4096*2
    float* ws_init  = ws_agg   + (size_t)NCH_ * 4096 * 2;  // NCH*4096*2
    int*   ws_rst   = (int*)(ws_init + (size_t)NCH_ * 4096 * 2);   // 64 ints
    __bf16* ws_wzb  = (__bf16*)((float*)ws_rst + 64);      // 64*8192 bf16 (1MB)
    __bf16* ws_zin  = ws_wzb + (size_t)HID_ * KZ_;         // T*8192 bf16 (64MB)

    float* out_main  = (float*)d_out;                      // (T, 128)
    float* out_state = out_main + (size_t)T_ * OUT_;       // (1,64,64) complex64

    k0_wzprep<<<(HID_ * KZ_) / 256, 256, 0, stream>>>(Wz, ws_wzb);

    k1_frontend<<<T_ / 64, 128, 0, stream>>>(
        x, W1, b1, W2, b2, Wgi, bgi, Wp, bp, Wgo, bgo, Ws, bs,
        ws_gated, ws_go, ws_skip);

    k2a_aggregate<<<NCH_ * 16, 256, 0, stream>>>(
        ws_gated, start, a, b, ws_agg, ws_rst);

    k2b_prefix<<<16, 256, 0, stream>>>(
        state0, ws_agg, ws_rst, a, b, ws_init, out_state);

    k2c_emit<<<NCH_ * 16, 256, 0, stream>>>(
        ws_gated, start, a, b, ws_init, ws_zin);

    k3_zgemm<<<T_ / 64, 128, 0, stream>>>(ws_zin, ws_wzb, bz, ws_z);

    k4_epilogue<<<T_ / 64, 128, 0, stream>>>(
        ws_z, ws_go, ws_skip, W3, b3, W4, b4, out_main);
}
